// HyperbolicLMHead_47639777247504
// MI455X (gfx1250) — matrix operand order repaired and checked by reading, not compiled, not run on hardware
//
#include <hip/hip_runtime.h>

// ---------------------------------------------------------------------------
// HyperbolicLMHead on MI455X (gfx1250, wave32, WMMA + Tensor Data Mover)
//   logits = sx @ sw^T - tx (outer) tw
//   sx/sw bf16 operands staged via TDM (tensor_load_to_lds), f32 accumulate.
// ---------------------------------------------------------------------------

typedef __attribute__((ext_vector_type(16))) __bf16       v16bf;
typedef __attribute__((ext_vector_type(8)))  __bf16       v8bf;
typedef __attribute__((ext_vector_type(4)))  __bf16       v4bf;
typedef __attribute__((ext_vector_type(8)))  float        v8f;
typedef __attribute__((ext_vector_type(4)))  unsigned int v4u;
typedef __attribute__((ext_vector_type(8)))  int          v8i;
typedef __attribute__((ext_vector_type(4)))  int          v4i;

#define DIM    1024
#define BATCH  4096
#define VOCAB  32000
#define BM     128
#define BN     128
#define BK     32            // K slab per LDS stage (1 WMMA K-step)
#define TILE_BYTES (BM * BK * 2)   // 8192 B per staged tile

// ---------------------------------------------------------------------------
// expmap0: row -> (cosh(||v||), sinh(||v||)/||v|| * v)   [bf16 space part]
// ---------------------------------------------------------------------------
__global__ __launch_bounds__(256)
void expmap_kernel(const float* __restrict__ in,
                   __bf16* __restrict__ s_out,
                   float*  __restrict__ t_out) {
    __shared__ float red[256];
    const int row = blockIdx.x;
    const int t   = threadIdx.x;

    const float4 x = ((const float4*)(in + (size_t)row * DIM))[t];
    red[t] = x.x * x.x + x.y * x.y + x.z * x.z + x.w * x.w;
    __syncthreads();
    #pragma unroll
    for (int s = 128; s > 0; s >>= 1) {
        if (t < s) red[t] += red[t + s];
        __syncthreads();
    }
    const float r    = sqrtf(red[0]);
    const float coef = sinhf(r) / fmaxf(r, 1e-15f);
    if (t == 0) t_out[row] = coshf(r);

    v4bf o;
    o.x = (__bf16)(coef * x.x);
    o.y = (__bf16)(coef * x.y);
    o.z = (__bf16)(coef * x.z);
    o.w = (__bf16)(coef * x.w);
    ((v4bf*)(s_out + (size_t)row * DIM))[t] = o;
}

// ---------------------------------------------------------------------------
// TDM: DMA one [128 rows x 32 K-elems] bf16 tile (row stride DIM) into LDS.
// D# packing per CDNA5 ISA §8.3/§8.4 (2D tensor, groups 2/3 zero).
// 6-arg builtin form (clang-23 / therock-10.0 headers).
// ---------------------------------------------------------------------------
__device__ __forceinline__ void tdm_tile_to_lds(const __bf16* gtile, unsigned lds_off) {
    const unsigned long long ga = (unsigned long long)(const void*)gtile;
    v4u g0;
    g0[0] = 1u;                                            // count=1 (user, load)
    g0[1] = lds_off;                                       // lds_addr [63:32]
    g0[2] = (unsigned)ga;                                  // global_addr[31:0]
    g0[3] = ((unsigned)(ga >> 32) & 0x01FFFFFFu)           // global_addr[56:32]
            | 0x80000000u;                                 // type=2 ("image")
    v8i g1;
    g1[0] = 0x00010000;                                    // data_size=1 (2 bytes)
    g1[1] = (int)(0x0400u << 16);                          // tensor_dim0=1024 (lo16)
    g1[2] = (int)(128u << 16);                             // dim0 hi16=0 | tensor_dim1=128 lo16
    g1[3] = (int)(32u << 16);                              // dim1 hi16=0 | tile_dim0=32
    g1[4] = 128;                                           // tile_dim1=128, tile_dim2=0
    g1[5] = 1024;                                          // tensor_dim0_stride lo32
    g1[6] = 0;                                             // stride hi16 | dim1_stride lo16
    g1[7] = 0;
    const v4i z4 = {0, 0, 0, 0};
    const v8i z8 = {0, 0, 0, 0, 0, 0, 0, 0};
    __builtin_amdgcn_tensor_load_to_lds(g0, g1, z4, z4, z8, 0);
}

// ---------------------------------------------------------------------------
// GEMM with rank-1 hyperbolic correction epilogue.
// Block tile 128x128, 8 waves (4 in M x 2 in N), wave tile 32x64 (2x4 WMMAs).
// Tiles staged by TDM, double-buffered; sync via TENSORcnt + barriers.
// ---------------------------------------------------------------------------
__global__ __launch_bounds__(256)
void hyp_gemm_kernel(const __bf16* __restrict__ A,   // [BATCH x DIM] bf16
                     const __bf16* __restrict__ B,   // [VOCAB x DIM] bf16
                     const float*  __restrict__ tx,  // [BATCH]
                     const float*  __restrict__ tw,  // [VOCAB]
                     float* __restrict__ out) {      // [BATCH x VOCAB]
    // layout: sA[0] sA[1] sB[0] sB[1] | sTx | sTw
    __shared__ __align__(128) char smem[4 * TILE_BYTES + 2 * 512];
    float* sTx = (float*)(smem + 4 * TILE_BYTES);
    float* sTw = (float*)(smem + 4 * TILE_BYTES + 512);

    const int tid   = threadIdx.x;
    const int lane  = tid & 31;
    const int l15   = lane & 15;
    const int half  = lane >> 4;                       // K-half select (ISA layout)
    const int wave  = __builtin_amdgcn_readfirstlane(tid >> 5);  // uniform
    const int waveM = wave & 3;                        // 4 waves along M
    const int waveN = wave >> 2;                       // 2 waves along N
    const int m0 = blockIdx.y * BM;
    const int n0 = blockIdx.x * BN;

    if (tid < BM)      sTx[tid]      = tx[m0 + tid];
    else               sTw[tid - BM] = tw[n0 + (tid - BM)];

    v8f acc[2][4];
    #pragma unroll
    for (int i = 0; i < 2; ++i)
        #pragma unroll
        for (int j = 0; j < 4; ++j)
            acc[i][j] = (v8f)0.0f;

    // LDS byte address = low 32 bits of flat shared address (aperture mapping)
    const unsigned lds_base = (unsigned)(unsigned long long)(size_t)(void*)smem;

    // prologue: DMA tile 0
    if (wave == 0) {
        tdm_tile_to_lds(A + (size_t)m0 * DIM, lds_base);
        tdm_tile_to_lds(B + (size_t)n0 * DIM, lds_base + 2 * TILE_BYTES);
    }

    const int NSTAGE = DIM / BK;   // 32
    for (int kb = 0; kb < NSTAGE; ++kb) {
        const int buf = kb & 1;

        if (wave == 0) {
            if (kb + 1 < NSTAGE) {
                const int k1 = (kb + 1) * BK;
                tdm_tile_to_lds(A + (size_t)m0 * DIM + k1,
                                lds_base + (1 - buf) * TILE_BYTES);
                tdm_tile_to_lds(B + (size_t)n0 * DIM + k1,
                                lds_base + 2 * TILE_BYTES + (1 - buf) * TILE_BYTES);
                __builtin_amdgcn_s_wait_tensorcnt(2);  // tile kb complete (in-order)
            } else {
                __builtin_amdgcn_s_wait_tensorcnt(0);  // drain last tile
            }
        }
        __syncthreads();   // buffer `buf` ready for all waves

        // ---- one K=32 WMMA step from LDS buffer `buf`
        const __bf16* a_base = (const __bf16*)(smem + buf * TILE_BYTES);
        const __bf16* b_base = (const __bf16*)(smem + (2 + buf) * TILE_BYTES);

        // Preload ALL fragments first (12x ds_load_b128), then one dscnt wait,
        // then an uninterrupted 8-WMMA burst.
        v16bf afrag[2];
        v16bf bfrag[4];
        // A 16x32 bf16: lanes 0-15 -> M=lane, K={0..7,16..23};
        //               lanes 16-31 -> M=lane-16, K={8..15,24..31}
        #pragma unroll
        for (int tm = 0; tm < 2; ++tm) {
            const __bf16* p = a_base + (waveM * 32 + tm * 16 + l15) * BK;
            v8bf lo = *(const v8bf*)(p + half * 8);
            v8bf hi = *(const v8bf*)(p + 16 + half * 8);
            afrag[tm] = __builtin_shufflevector(lo, hi,
                0, 1, 2, 3, 4, 5, 6, 7, 8, 9, 10, 11, 12, 13, 14, 15);
        }
        // B 32x16 bf16: lanes 0-15 -> N=lane, K={0..7,16..23}; lanes 16-31 -> K={8..15,24..31}
        #pragma unroll
        for (int tn = 0; tn < 4; ++tn) {
            const __bf16* p = b_base + (waveN * 64 + tn * 16 + l15) * BK;
            v8bf lo = *(const v8bf*)(p + half * 8);
            v8bf hi = *(const v8bf*)(p + 16 + half * 8);
            bfrag[tn] = __builtin_shufflevector(lo, hi,
                0, 1, 2, 3, 4, 5, 6, 7, 8, 9, 10, 11, 12, 13, 14, 15);
        }
        #pragma unroll
        for (int tn = 0; tn < 4; ++tn)
            #pragma unroll
            for (int tm = 0; tm < 2; ++tm)
                acc[tm][tn] = __builtin_amdgcn_wmma_f32_16x16x32_bf16(
                    false, afrag[tm], false, bfrag[tn],
                    (short)0, acc[tm][tn], false, false);

        __syncthreads();   // all waves done reading `buf` before TDM reuses it
    }

    // --- epilogue: D[m,n] = acc - tx[m]*tw[n]
    // C/D layout: lanes 0-15 -> N=lane, VGPR r -> M=r; lanes 16-31 -> M=r+8
    // Output is write-once streaming (524 MB >> L2): use non-temporal stores
    // so the bf16 operand working set stays resident in the 192 MB L2.
    #pragma unroll
    for (int tm = 0; tm < 2; ++tm) {
        #pragma unroll
        for (int tn = 0; tn < 4; ++tn) {
            const int nl = waveN * 64 + tn * 16 + l15;
            const float twv = sTw[nl];
            float* orow = out + (size_t)n0 + nl;
            #pragma unroll
            for (int r = 0; r < 8; ++r) {
                const int ml = waveM * 32 + tm * 16 + r + half * 8;
                __builtin_nontemporal_store(acc[tm][tn][r] - sTx[ml] * twv,
                                            &orow[(size_t)(m0 + ml) * VOCAB]);
            }
        }
    }
}

// ---------------------------------------------------------------------------
// launch
// ---------------------------------------------------------------------------
extern "C" void kernel_launch(void* const* d_in, const int* in_sizes, int n_in,
                              void* d_out, int out_size, void* d_ws, size_t ws_size,
                              hipStream_t stream) {
    const float* x = (const float*)d_in[0];   // [4096 x 1024]
    const float* W = (const float*)d_in[1];   // [32000 x 1024]
    float* out = (float*)d_out;               // [4096 x 32000]

    char* ws = (char*)d_ws;
    __bf16* sx = (__bf16*)ws;                                   //  8,388,608 B
    __bf16* sw = (__bf16*)(ws + (size_t)BATCH * DIM * 2);       // 65,536,000 B
    float*  tx = (float*)(ws + (size_t)(BATCH + VOCAB) * DIM * 2);
    float*  tw = tx + BATCH;

    expmap_kernel<<<BATCH, 256, 0, stream>>>(x, sx, tx);
    expmap_kernel<<<VOCAB, 256, 0, stream>>>(W, sw, tw);

    dim3 grid(VOCAB / BN, BATCH / BM);        // (250, 32)
    hyp_gemm_kernel<<<grid, 256, 0, stream>>>(sx, sw, tx, tw, out);
}
